// MultiheadAttention_44573170598137
// MI455X (gfx1250) — matrix operand order repaired and checked by reading, not compiled, not run on hardware
//
#include <hip/hip_runtime.h>

#define DEV static __device__ __forceinline__
typedef unsigned short u16;
typedef unsigned int   u32;
typedef __attribute__((ext_vector_type(16))) __bf16 v16bf;
typedef __attribute__((ext_vector_type(8)))  float  v8f;
typedef __attribute__((ext_vector_type(2)))  __bf16 v2bf;
typedef int v4i __attribute__((vector_size(16)));

#define AS1 __attribute__((address_space(1)))
#define AS3 __attribute__((address_space(3)))

#if defined(__has_builtin)
#if __has_builtin(__builtin_amdgcn_global_load_async_to_lds_b128)
#define HAS_ASYNC 1
#endif
#if __has_builtin(__builtin_amdgcn_sched_barrier)
#define SCHED_BARRIER() __builtin_amdgcn_sched_barrier(0)
#endif
#endif
#ifndef HAS_ASYNC
#define HAS_ASYNC 0
#endif
#ifndef SCHED_BARRIER
#define SCHED_BARRIER() ((void)0)
#endif

#if HAS_ASYNC
#if __has_builtin(__builtin_amdgcn_s_wait_asynccnt)
#define ASYNC_WAIT() __builtin_amdgcn_s_wait_asynccnt(0)
#else
#define ASYNC_WAIT() asm volatile("s_wait_asynccnt 0" ::: "memory")
#endif
#else
#define ASYNC_WAIT() ((void)0)
#endif

union AFrag { v16bf v; uint4 q[2]; };

DEV u16 f2bf(float f) {
  __bf16 h = (__bf16)f;                    // backend: v_cvt_pk_bf16_f32 path
  return __builtin_bit_cast(u16, h);
}
DEV u32 pack2(float a, float b) {
  v2bf v; v[0] = (__bf16)a; v[1] = (__bf16)b;
  return __builtin_bit_cast(u32, v);
}
DEV uint4 packf4x2(float4 a, float4 b) {
  uint4 r; r.x = pack2(a.x, a.y); r.y = pack2(a.z, a.w);
  r.z = pack2(b.x, b.y); r.w = pack2(b.z, b.w); return r;
}
DEV v8f wmma_bf16(v16bf a, v16bf b, v8f c) {
  return __builtin_amdgcn_wmma_f32_16x16x32_bf16(false, a, false, b, (short)0, c, false, false);
}

// 16-byte global -> LDS copy; OFF bytes applied to both sides (IOFFSET).
template <int OFF>
DEV void async_cp16(void* lds, const void* g) {
#if HAS_ASYNC
  __builtin_amdgcn_global_load_async_to_lds_b128(
      (AS1 v4i*)g, (AS3 v4i*)lds, OFF, 0);
#else
  *(uint4*)((char*)lds + OFF) = *(const uint4*)((const char*)g + OFF);
#endif
}

// ---------------------------------------------------------------------------
// One-shot f32 -> bf16 weight conversion (1M elements per matrix).
// ---------------------------------------------------------------------------
__global__ __launch_bounds__(256) void cvt_kernel(
    const float* __restrict__ src, u16* __restrict__ dst)
{
  const int idx = (blockIdx.x * 256 + threadIdx.x) * 8;
  float4 a = *(const float4*)(src + idx);
  float4 b = *(const float4*)(src + idx + 4);
  *(uint4*)(dst + idx) = packf4x2(a, b);
}

// ---------------------------------------------------------------------------
// QKV projection: Y = (X @ W^T + bias) [* 0.125 for Q], K-chunk = 64.
// X f32 rows r = b*1024 + t; W pre-converted bf16.
// mode 0: Q -> out[((b*16+h)*1024 + t)*64 + d]  (scaled)
// mode 1: K -> out[((b*16+h)*1024 + s)*64 + d]
// mode 2: V -> out[((b*16+h)*64 + d)*1024 + s]  (transposed, packed stores)
// ---------------------------------------------------------------------------
__global__ __launch_bounds__(256) void qkv_proj_kernel(
    const float* __restrict__ X, const u16* __restrict__ Wb,
    const float* __restrict__ bias, u16* __restrict__ out, int mode)
{
  __shared__ __align__(16) float aTf[2][128 * 64];   // f32 staging, cvt at frag build
  __shared__ __align__(16) u16   bT[2][64 * 64];     // bf16 weights
  const v8f vzero = {0.f,0.f,0.f,0.f,0.f,0.f,0.f,0.f};

  const int tid = threadIdx.x;
  const int w = tid >> 5, lane = tid & 31, hi = lane >> 4, ln = lane & 15;
  const int m0 = blockIdx.x * 128, n0 = blockIdx.y * 64;

  const int arow = tid >> 1, acol = (tid & 1) * 32;  // float units, 128B each
  const int brow = tid >> 2, bcol = (tid & 3) * 16;  // u16 units, 32B each
  const int ra = m0 + arow;                          // r = b*1024 + t
  const float* gA = X + ((size_t)((ra & 1023) * 8 + (ra >> 10))) * 1024 + acol;
  const u16*   gB = Wb + (size_t)(n0 + brow) * 1024 + bcol;

  // prefetch k-chunk 0
  {
    float* aL = &aTf[0][arow * 64 + acol];
    async_cp16<  0>(aL, gA); async_cp16< 16>(aL, gA);
    async_cp16< 32>(aL, gA); async_cp16< 48>(aL, gA);
    async_cp16< 64>(aL, gA); async_cp16< 80>(aL, gA);
    async_cp16< 96>(aL, gA); async_cp16<112>(aL, gA);
    u16* bL = &bT[0][brow * 64 + bcol];
    async_cp16<0>(bL, gB); async_cp16<16>(bL, gB);
  }

  v8f acc[4];
  #pragma unroll
  for (int j = 0; j < 4; j++) acc[j] = vzero;

  for (int kt = 0; kt < 16; kt++) {
    const int buf = kt & 1;
    ASYNC_WAIT();
    __syncthreads();
    if (kt + 1 < 16) {
      const int kb2 = (kt + 1) * 64;
      float* aL = &aTf[buf ^ 1][arow * 64 + acol];
      const float* gA2 = gA + kb2;
      async_cp16<  0>(aL, gA2); async_cp16< 16>(aL, gA2);
      async_cp16< 32>(aL, gA2); async_cp16< 48>(aL, gA2);
      async_cp16< 64>(aL, gA2); async_cp16< 80>(aL, gA2);
      async_cp16< 96>(aL, gA2); async_cp16<112>(aL, gA2);
      u16* bL = &bT[buf ^ 1][brow * 64 + bcol];
      const u16* gB2 = gB + kb2;
      async_cp16<0>(bL, gB2); async_cp16<16>(bL, gB2);
    }
    #pragma unroll
    for (int c = 0; c < 2; c++) {
      // A fragment: f32 from LDS, convert to bf16
      const float* ap = &aTf[buf][(w * 16 + ln) * 64 + c * 32 + hi * 8];
      float4 f0 = *(const float4*)(ap);
      float4 f1 = *(const float4*)(ap + 4);
      float4 f2 = *(const float4*)(ap + 16);
      float4 f3 = *(const float4*)(ap + 20);
      AFrag a; a.q[0] = packf4x2(f0, f1); a.q[1] = packf4x2(f2, f3);
      // B fragments: issue all 8 ds_loads, then 4 back-to-back WMMAs
      AFrag bf[4];
      #pragma unroll
      for (int j = 0; j < 4; j++) {
        const int base = (j * 16 + ln) * 64 + c * 32 + hi * 8;
        bf[j].q[0] = *(const uint4*)&bT[buf][base];
        bf[j].q[1] = *(const uint4*)&bT[buf][base + 16];
      }
      SCHED_BARRIER();
      #pragma unroll
      for (int j = 0; j < 4; j++) acc[j] = wmma_bf16(a.v, bf[j].v, acc[j]);
    }
  }

  const int r0 = m0 + w * 16 + hi * 8;
  #pragma unroll
  for (int j = 0; j < 4; j++) {
    const int n = n0 + j * 16 + ln;
    const int h = n >> 6, d = n & 63;
    const float bv = bias[n];
    if (mode == 2) {
      const int b = r0 >> 10, s = r0 & 1023;    // 8 consecutive s values
      uint4 st;
      st.x = pack2(acc[j][0] + bv, acc[j][1] + bv);
      st.y = pack2(acc[j][2] + bv, acc[j][3] + bv);
      st.z = pack2(acc[j][4] + bv, acc[j][5] + bv);
      st.w = pack2(acc[j][6] + bv, acc[j][7] + bv);
      *(uint4*)&out[(((size_t)(b * 16 + h)) * 64 + d) * 1024 + s] = st;
    } else {
      const float scale = (mode == 0) ? 0.125f : 1.0f;
      #pragma unroll
      for (int i = 0; i < 8; i++) {
        const int r = r0 + i;
        const int b = r >> 10, t = r & 1023;
        out[(((size_t)(b * 16 + h)) * 1024 + t) * 64 + d] =
            f2bf((acc[j][i] + bv) * scale);
      }
    }
  }
}

// ---------------------------------------------------------------------------
// Flash attention: per (b, h, 128-row Q block). S streamed in 64-chunks,
// K/V tiles double-buffered via async-to-LDS.
// Q: (b,h,t,64) bf16 (pre-scaled), K: (b,h,s,64) bf16, Vt: (b,h,d,s) bf16.
// Output attn: (b*1024+t, 1024) bf16, column = h*64 + d.
// ---------------------------------------------------------------------------
__global__ __launch_bounds__(256) void attn_kernel(
    const u16* __restrict__ Q, const u16* __restrict__ K,
    const u16* __restrict__ Vt, const int* __restrict__ mask,
    u16* __restrict__ attn)
{
  __shared__ __align__(16) u16 qT[128 * 64];
  __shared__ __align__(16) u16 kT[2][64 * 64];
  __shared__ __align__(16) u16 vT[2][64 * 64];
  __shared__ __align__(16) u16 pT[8 * 16 * 64];
  const v8f vzero = {0.f,0.f,0.f,0.f,0.f,0.f,0.f,0.f};

  const int tid = threadIdx.x;
  const int w = tid >> 5, lane = tid & 31, hi = lane >> 4, ln = lane & 15;
  const int qb = blockIdx.x, h = blockIdx.y, b = blockIdx.z;
  const size_t bh = (size_t)(b * 16 + h);
  const u16* qG = Q  + bh * 65536 + (size_t)qb * 8192;
  const u16* kG = K  + bh * 65536;
  const u16* vG = Vt + bh * 65536;
  const int vr = tid >> 2, vc = (tid & 3) * 16;

  {   // initial copies: Q tile + S-chunk 0
    u16* qL = &qT[tid * 32];
    const u16* qg = qG + tid * 32;
    async_cp16< 0>(qL, qg); async_cp16<16>(qL, qg);
    async_cp16<32>(qL, qg); async_cp16<48>(qL, qg);
    u16* kL = &kT[0][tid * 16];
    const u16* kg = kG + tid * 16;
    async_cp16<0>(kL, kg); async_cp16<16>(kL, kg);
    u16* vL = &vT[0][vr * 64 + vc];
    const u16* vg = vG + (size_t)vr * 1024 + vc;
    async_cp16<0>(vL, vg); async_cp16<16>(vL, vg);
  }
  ASYNC_WAIT();
  __syncthreads();

  AFrag qA[2];
  #pragma unroll
  for (int c = 0; c < 2; c++) {
    qA[c].q[0] = *(const uint4*)&qT[(w * 16 + ln) * 64 + c * 32 + hi * 8];
    qA[c].q[1] = *(const uint4*)&qT[(w * 16 + ln) * 64 + c * 32 + hi * 8 + 16];
  }

  float m_i[8], l_i[8];
  v8f accO[4];
  #pragma unroll
  for (int i = 0; i < 8; i++) { m_i[i] = -INFINITY; l_i[i] = 0.0f; }
  #pragma unroll
  for (int j = 0; j < 4; j++) accO[j] = vzero;

  const int pB = w * 1024;   // wave-private 16x64 P tile

  for (int it = 0; it < 16; it++) {
    const int s0 = it * 64, buf = it & 1;
    if (it) { ASYNC_WAIT(); __syncthreads(); }
    if (it + 1 < 16) {
      const int s1 = s0 + 64;
      u16* kL = &kT[buf ^ 1][tid * 16];
      const u16* kg = kG + s1 * 64 + tid * 16;
      async_cp16<0>(kL, kg); async_cp16<16>(kL, kg);
      u16* vL = &vT[buf ^ 1][vr * 64 + vc];
      const u16* vg = vG + (size_t)vr * 1024 + s1 + vc;
      async_cp16<0>(vL, vg); async_cp16<16>(vL, vg);
    }

    // scores = Q Kᵀ  (contract over d = 64 in two 32-chunks)
    v8f sc[4];
    #pragma unroll
    for (int j = 0; j < 4; j++) sc[j] = vzero;
    #pragma unroll
    for (int c = 0; c < 2; c++) {
      AFrag kf[4];
      #pragma unroll
      for (int j = 0; j < 4; j++) {
        const int base = (j * 16 + ln) * 64 + c * 32 + hi * 8;
        kf[j].q[0] = *(const uint4*)&kT[buf][base];
        kf[j].q[1] = *(const uint4*)&kT[buf][base + 16];
      }
      SCHED_BARRIER();
      #pragma unroll
      for (int j = 0; j < 4; j++) sc[j] = wmma_bf16(qA[c].v, kf[j].v, sc[j]);
    }

    // key padding mask: scores -= FLT_MAX * mask[s]
    #pragma unroll
    for (int j = 0; j < 4; j++) {
      const float mf = (float)mask[b * 1024 + s0 + j * 16 + ln] * 3.402823466e38f;
      #pragma unroll
      for (int i = 0; i < 8; i++) sc[j][i] -= mf;
    }

    // online softmax; row = (vgpr i, lane half), 16 lanes wide
    #pragma unroll
    for (int i = 0; i < 8; i++) {
      float rm = fmaxf(fmaxf(sc[0][i], sc[1][i]), fmaxf(sc[2][i], sc[3][i]));
      rm = fmaxf(rm, __shfl_xor(rm, 1));
      rm = fmaxf(rm, __shfl_xor(rm, 2));
      rm = fmaxf(rm, __shfl_xor(rm, 4));
      rm = fmaxf(rm, __shfl_xor(rm, 8));
      const float mn = fmaxf(m_i[i], rm);
      const float corr = __expf(m_i[i] - mn);
      float rs = 0.0f;
      #pragma unroll
      for (int j = 0; j < 4; j++) {
        const float p = __expf(sc[j][i] - mn);
        rs += p;
        pT[pB + (hi * 8 + i) * 64 + j * 16 + ln] = f2bf(p);
      }
      rs += __shfl_xor(rs, 1);
      rs += __shfl_xor(rs, 2);
      rs += __shfl_xor(rs, 4);
      rs += __shfl_xor(rs, 8);
      l_i[i] = l_i[i] * corr + rs;
      m_i[i] = mn;
      #pragma unroll
      for (int j = 0; j < 4; j++) accO[j][i] *= corr;
    }

    // cross-lane LDS round-trip within the wave: wait for DS stores
    asm volatile("s_wait_dscnt 0" ::: "memory");

    // accO += P V  (contract over s = 64 in two 32-chunks)
    AFrag pA[2];
    #pragma unroll
    for (int c = 0; c < 2; c++) {
      pA[c].q[0] = *(const uint4*)&pT[pB + ln * 64 + c * 32 + hi * 8];
      pA[c].q[1] = *(const uint4*)&pT[pB + ln * 64 + c * 32 + hi * 8 + 16];
    }
    #pragma unroll
    for (int c = 0; c < 2; c++) {
      AFrag vb[4];
      #pragma unroll
      for (int j = 0; j < 4; j++) {
        const int base = (j * 16 + ln) * 64 + c * 32 + hi * 8;
        vb[j].q[0] = *(const uint4*)&vT[buf][base];
        vb[j].q[1] = *(const uint4*)&vT[buf][base + 16];
      }
      SCHED_BARRIER();
      #pragma unroll
      for (int j = 0; j < 4; j++) accO[j] = wmma_bf16(pA[c].v, vb[j].v, accO[j]);
    }
  }

  const int t0 = qb * 128 + w * 16 + hi * 8;
  #pragma unroll
  for (int i = 0; i < 8; i++) {
    const float inv = 1.0f / l_i[i];
    #pragma unroll
    for (int j = 0; j < 4; j++) {
      const int col = h * 64 + j * 16 + ln;
      attn[(size_t)(b * 1024 + t0 + i) * 1024 + col] = f2bf(accO[j][i] * inv);
    }
  }
}

// ---------------------------------------------------------------------------
// Output projection: out = attn @ Wo^T + bo, fp32 result in (T, B, D) order.
// attn rows r = b*1024 + t (row-major bf16); Wo pre-converted bf16. K-chunk 64.
// ---------------------------------------------------------------------------
__global__ __launch_bounds__(256) void out_proj_kernel(
    const u16* __restrict__ A, const u16* __restrict__ Wb,
    const float* __restrict__ bias, float* __restrict__ out)
{
  __shared__ __align__(16) u16 aT[2][128 * 64];
  __shared__ __align__(16) u16 bT[2][64 * 64];
  const v8f vzero = {0.f,0.f,0.f,0.f,0.f,0.f,0.f,0.f};

  const int tid = threadIdx.x;
  const int w = tid >> 5, lane = tid & 31, hi = lane >> 4, ln = lane & 15;
  const int m0 = blockIdx.x * 128, n0 = blockIdx.y * 64;

  const int arow = tid >> 1, acol = (tid & 1) * 32;  // u16 units, 64B each
  const int brow = tid >> 2, bcol = (tid & 3) * 16;  // u16 units, 32B each
  const u16* gA = A  + (size_t)(m0 + arow) * 1024 + acol;
  const u16* gB = Wb + (size_t)(n0 + brow) * 1024 + bcol;

  {
    u16* aL = &aT[0][arow * 64 + acol];
    async_cp16< 0>(aL, gA); async_cp16<16>(aL, gA);
    async_cp16<32>(aL, gA); async_cp16<48>(aL, gA);
    u16* bL = &bT[0][brow * 64 + bcol];
    async_cp16<0>(bL, gB); async_cp16<16>(bL, gB);
  }

  v8f acc[4];
  #pragma unroll
  for (int j = 0; j < 4; j++) acc[j] = vzero;

  for (int kt = 0; kt < 16; kt++) {
    const int buf = kt & 1;
    ASYNC_WAIT();
    __syncthreads();
    if (kt + 1 < 16) {
      const int kb2 = (kt + 1) * 64;
      u16* aL = &aT[buf ^ 1][arow * 64 + acol];
      const u16* gA2 = gA + kb2;
      async_cp16< 0>(aL, gA2); async_cp16<16>(aL, gA2);
      async_cp16<32>(aL, gA2); async_cp16<48>(aL, gA2);
      u16* bL = &bT[buf ^ 1][brow * 64 + bcol];
      const u16* gB2 = gB + kb2;
      async_cp16<0>(bL, gB2); async_cp16<16>(bL, gB2);
    }
    #pragma unroll
    for (int c = 0; c < 2; c++) {
      AFrag a;
      const int abase = (w * 16 + ln) * 64 + c * 32 + hi * 8;
      a.q[0] = *(const uint4*)&aT[buf][abase];
      a.q[1] = *(const uint4*)&aT[buf][abase + 16];
      AFrag bf[4];
      #pragma unroll
      for (int j = 0; j < 4; j++) {
        const int base = (j * 16 + ln) * 64 + c * 32 + hi * 8;
        bf[j].q[0] = *(const uint4*)&bT[buf][base];
        bf[j].q[1] = *(const uint4*)&bT[buf][base + 16];
      }
      SCHED_BARRIER();
      #pragma unroll
      for (int j = 0; j < 4; j++) acc[j] = wmma_bf16(a.v, bf[j].v, acc[j]);
    }
  }

  const int r0 = m0 + w * 16 + hi * 8;
  #pragma unroll
  for (int j = 0; j < 4; j++) {
    const int n = n0 + j * 16 + ln;
    const float bv = bias[n];
    #pragma unroll
    for (int i = 0; i < 8; i++) {
      const int r = r0 + i;
      const int b = r >> 10, t = r & 1023;
      out[((size_t)(t * 8 + b)) * 1024 + n] = acc[j][i] + bv;
    }
  }
}

// ---------------------------------------------------------------------------
extern "C" void kernel_launch(void* const* d_in, const int* in_sizes, int n_in,
                              void* d_out, int out_size, void* d_ws, size_t ws_size,
                              hipStream_t stream) {
  const float* Xq   = (const float*)d_in[0];
  const float* Xk   = (const float*)d_in[1];
  const float* Xv   = (const float*)d_in[2];
  const int*   mask = (const int*)  d_in[3];
  const float* Wq   = (const float*)d_in[4];
  const float* Wk   = (const float*)d_in[5];
  const float* Wv   = (const float*)d_in[6];
  const float* Wo   = (const float*)d_in[7];
  const float* bq   = (const float*)d_in[8];
  const float* bk   = (const float*)d_in[9];
  const float* bv   = (const float*)d_in[10];
  const float* bo   = (const float*)d_in[11];

  const size_t N = 8ull * 1024 * 1024;   // elements per bf16 intermediate
  const size_t M = 1024ull * 1024;       // elements per weight matrix
  u16* qw  = (u16*)d_ws;
  u16* kw  = qw + N;
  u16* vw  = kw + N;
  u16* aw  = vw + N;
  u16* wqb = aw + N;
  u16* wkb = wqb + M;
  u16* wvb = wkb + M;
  u16* wob = wvb + M;

  dim3 blk(256);
  dim3 gProj(64, 16);
  cvt_kernel<<<512, blk, 0, stream>>>(Wq, wqb);
  cvt_kernel<<<512, blk, 0, stream>>>(Wk, wkb);
  cvt_kernel<<<512, blk, 0, stream>>>(Wv, wvb);
  cvt_kernel<<<512, blk, 0, stream>>>(Wo, wob);
  qkv_proj_kernel<<<gProj, blk, 0, stream>>>(Xq, wqb, bq, qw, 0);
  qkv_proj_kernel<<<gProj, blk, 0, stream>>>(Xk, wkb, bk, kw, 1);
  qkv_proj_kernel<<<gProj, blk, 0, stream>>>(Xv, wvb, bv, vw, 2);
  attn_kernel<<<dim3(8, 16, 8), blk, 0, stream>>>(qw, kw, vw, mask, aw);
  out_proj_kernel<<<gProj, blk, 0, stream>>>(aw, wob, bo, (float*)d_out);
}
